// FullAttention_21612275434076
// MI455X (gfx1250) — matrix operand order repaired and checked by reading, not compiled, not run on hardware
//
#include <hip/hip_runtime.h>

typedef __attribute__((ext_vector_type(16))) _Float16 v16h;
typedef __attribute__((ext_vector_type(2)))  _Float16 v2h;
typedef __attribute__((ext_vector_type(8)))  float    v8f;
typedef int v4i_vs __attribute__((vector_size(16)));   // matches builtin param type

// Problem dims (fixed by reference)
constexpr int Bb = 2, Ll = 2048, Hh = 16, Ee = 64;

constexpr int BR = 128;                // query rows per block (8 waves x 16)
constexpr int BC = 32;                 // keys per inner tile
constexpr int NWAVE = BR / 16;
constexpr int NTHREADS = NWAVE * 32;   // 256 threads, wave32

// Row pads keep every fragment run 16B-aligned while skewing banks
constexpr int KPAD = 8;   // halves: row stride 72 halves = 144B (16B multiple)
constexpr int VPAD = 8;   // halves: row stride 40 halves =  80B (16B multiple)

union HU { unsigned u[8]; v16h h; };

// ---------- CDNA5 async global->LDS path (probed via __has_builtin) ----------
#if defined(__has_builtin)
#  if __has_builtin(__builtin_amdgcn_global_load_async_to_lds_b128)
#    define HAVE_ASYNC_LDS 1
#  endif
#endif

#if defined(HAVE_ASYNC_LDS)
__device__ __forceinline__ void async_cp16(const void* g, void* l) {
    __builtin_amdgcn_global_load_async_to_lds_b128(
        (__attribute__((address_space(1))) v4i_vs*)g,
        (__attribute__((address_space(3))) v4i_vs*)l,
        /*imm offset*/0, /*cpol*/0);
}
#  if __has_builtin(__builtin_amdgcn_s_wait_asynccnt)
#    define WAIT_ASYNC(n) __builtin_amdgcn_s_wait_asynccnt(n)
#  else
#    define WAIT_ASYNC(n) asm volatile("s_wait_asynccnt " #n ::: "memory")
#  endif
#endif

// =====================================================================
// Pre-pass: K [B,L,H,E] f32 -> Kh [B,H,L,E] f16
//           V [B,L,H,E] f32 -> VTh [B,H,E,L] f16  (transposed via LDS tile)
// =====================================================================
__global__ __launch_bounds__(256)
void cvt_kv_kernel(const float* __restrict__ Kg, const float* __restrict__ Vg,
                   _Float16* __restrict__ Kh, _Float16* __restrict__ VTh)
{
    __shared__ __align__(16) _Float16 sT[64][72];   // [l][e], padded rows

    const int b = blockIdx.z, h = blockIdx.y, l0 = blockIdx.x * 64;
    const int tid = threadIdx.x;
    const size_t rowStride = (size_t)Hh * Ee;

    // coalesced read of a 64(l) x 64(e) tile; direct write of Kh
    {
        const int lr = tid >> 2;            // 0..63
        const int cb = (tid & 3) * 16;      // 0,16,32,48
        const float* ksrc = Kg + ((size_t)b * Ll + l0 + lr) * rowStride + (size_t)h * Ee + cb;
        const float* vsrc = Vg + ((size_t)b * Ll + l0 + lr) * rowStride + (size_t)h * Ee + cb;
        _Float16* kdst = Kh + (((size_t)(b * Hh + h) * Ll) + l0 + lr) * Ee + cb;
        #pragma unroll
        for (int i = 0; i < 16; ++i) {
            kdst[i]        = (_Float16)ksrc[i];
            sT[lr][cb + i] = (_Float16)vsrc[i];
        }
    }
    __syncthreads();
    // transposed, coalesced write of VTh
    {
        const int er = tid >> 2;            // 0..63 (E row)
        const int lc = (tid & 3) * 16;      // 0,16,32,48
        _Float16* vdst = VTh + ((size_t)(b * Hh + h) * Ee + er) * Ll + l0 + lc;
        #pragma unroll
        for (int i = 0; i < 16; ++i)
            vdst[i] = sT[lc + i][er];
    }
}

// =====================================================================
// Main kernel: flash attention on pre-converted f16 K / V^T, S^T/O^T
// formulation, double-buffered (async if available) LDS staging.
// =====================================================================
__device__ __forceinline__ void stage_tile(const _Float16* __restrict__ Khb,
                                           const _Float16* __restrict__ VThb,
                                           int k0, int tid,
                                           _Float16 (*sKb)[Ee + KPAD],
                                           _Float16 (*sVb)[BC + VPAD])
{
    // K portion: 32 keys x 64 E, 16B per thread
    const int r = tid >> 3, c = tid & 7;
    const _Float16* ksrc = Khb + (size_t)(k0 + r) * Ee + c * 8;
    _Float16*       kdst = &sKb[r][c * 8];
    // V portion: 64 E x 32 keys, 16B per thread
    const int e = tid >> 2, kc = tid & 3;
    const _Float16* vsrc = VThb + (size_t)e * Ll + k0 + kc * 8;
    _Float16*       vdst = &sVb[e][kc * 8];
#if defined(HAVE_ASYNC_LDS)
    async_cp16(ksrc, kdst);
    async_cp16(vsrc, vdst);
#else
    *(uint4*)kdst = *(const uint4*)ksrc;
    *(uint4*)vdst = *(const uint4*)vsrc;
#endif
}

__global__ __launch_bounds__(NTHREADS)
void fa_causal_wmma_split(const float* __restrict__ Qg,
                          const _Float16* __restrict__ Kh,
                          const _Float16* __restrict__ VTh,
                          float* __restrict__ Og)
{
    __shared__ __align__(16) _Float16 sK[2][BC][Ee + KPAD];    // [buf][key][e]
    __shared__ __align__(16) _Float16 sVT[2][Ee][BC + VPAD];   // [buf][e][key]

    const int b  = blockIdx.z;
    const int h  = blockIdx.y;
    const int q0 = blockIdx.x * BR;

    const int tid  = threadIdx.x;
    const int wave = tid >> 5;
    const int lane = tid & 31;
    const int l16  = lane & 15;
    const int hi   = lane >> 4;

    const size_t rowStride = (size_t)Hh * Ee;
    const float*    Qb   = Qg + ((size_t)b * Ll) * rowStride + (size_t)h * Ee;
    float*          Ob   = Og + ((size_t)b * Ll) * rowStride + (size_t)h * Ee;
    const _Float16* Khb  = Kh  + (size_t)(b * Hh + h) * Ll * Ee;
    const _Float16* VThb = VTh + (size_t)(b * Hh + h) * Ee * Ll;

    const int qrow_base = q0 + wave * 16;
    const int myq       = qrow_base + l16;

    // Q as B-fragments of Q^T: chunk c, elem e -> E = 32c + 8hi + (e&7) + 16(e>>3)
    v16h qb[2];
    {
        const float* qrow = Qb + (size_t)myq * rowStride;
        #pragma unroll
        for (int c = 0; c < 2; ++c)
            #pragma unroll
            for (int e = 0; e < 16; ++e)
                qb[c][e] = (_Float16)qrow[32 * c + 8 * hi + (e & 7) + 16 * (e >> 3)];
    }

    v8f o[4] = {};
    float m = -INFINITY, lsum = 0.0f;

    const int jmax = (q0 + BR + BC - 1) / BC;

    stage_tile(Khb, VThb, 0, tid, sK[0], sVT[0]);     // prologue: tile 0 -> buf 0

    for (int jt = 0; jt < jmax; ++jt) {
        const int k0  = jt * BC;
        const int buf = jt & 1;

        __syncthreads();   // all waves done computing tile jt-1 (frees buf (jt+1)&1)

        if (jt + 1 < jmax)
            stage_tile(Khb, VThb, k0 + BC, tid, sK[buf ^ 1], sVT[buf ^ 1]);

#if defined(HAVE_ASYNC_LDS)
        if (jt + 1 < jmax) { WAIT_ASYNC(2); }   // tile jt arrived (in-order per wave)
        else              { WAIT_ASYNC(0); }
#endif
        __syncthreads();   // all waves' portions of tile jt visible

        // Uniform causal skip (scalar branch; EXEC stays all-ones around WMMA)
        if (__builtin_amdgcn_readfirstlane((int)(k0 <= qrow_base + 15))) {

            // ---- S^T = K_tile @ Q^T ; frag j covers keys 16j..16j+15 ----
            v8f s[2];
            #pragma unroll
            for (int j = 0; j < 2; ++j) {
                v16h ka0, ka1;
                #pragma unroll
                for (int e = 0; e < 8; ++e) {
                    ka0[e]     = sK[buf][16 * j + l16][8 * hi + e];
                    ka0[8 + e] = sK[buf][16 * j + l16][16 + 8 * hi + e];
                    ka1[e]     = sK[buf][16 * j + l16][32 + 8 * hi + e];
                    ka1[8 + e] = sK[buf][16 * j + l16][48 + 8 * hi + e];
                }
                v8f acc = {};
                acc = __builtin_amdgcn_wmma_f32_16x16x32_f16(false, ka0, false, qb[0],
                                                             (short)0, acc, false, false);
                acc = __builtin_amdgcn_wmma_f32_16x16x32_f16(false, ka1, false, qb[1],
                                                             (short)0, acc, false, false);
                s[j] = acc;
            }

            // ---- Causal mask: key = k0 + 16j + r + 8hi, query = myq ----
            if (k0 + BC - 1 > qrow_base) {
                #pragma unroll
                for (int j = 0; j < 2; ++j)
                    #pragma unroll
                    for (int r = 0; r < 8; ++r)
                        if (k0 + 16 * j + 8 * hi + r > myq) s[j][r] = -INFINITY;
            }

            // ---- Online softmax: one query per lane ----
            float rmax = fmaxf(s[0][0], s[1][0]);
            #pragma unroll
            for (int r = 1; r < 8; ++r)
                rmax = fmaxf(rmax, fmaxf(s[0][r], s[1][r]));
            rmax = fmaxf(rmax, __shfl_xor(rmax, 16));

            const float mnew = fmaxf(m, rmax);
            const float corr = __expf(m - mnew);
            m = mnew;

            float rs = 0.0f;
            #pragma unroll
            for (int j = 0; j < 2; ++j)
                #pragma unroll
                for (int r = 0; r < 8; ++r) {
                    s[j][r] = __expf(s[j][r] - mnew);
                    rs += s[j][r];
                }
            rs += __shfl_xor(rs, 16);
            lsum = lsum * corr + rs;

            #pragma unroll
            for (int t = 0; t < 4; ++t)
                #pragma unroll
                for (int r = 0; r < 8; ++r)
                    o[t][r] *= corr;

            // ---- Build P^T B-fragment in registers (pack f16 + half-swap) ----
            unsigned pk[2][4], xk[2][4];
            #pragma unroll
            for (int j = 0; j < 2; ++j)
                #pragma unroll
                for (int q = 0; q < 4; ++q) {
                    v2h t;
                    t[0] = (_Float16)s[j][2 * q];
                    t[1] = (_Float16)s[j][2 * q + 1];
                    pk[j][q] = __builtin_bit_cast(unsigned, t);
                    xk[j][q] = (unsigned)__shfl_xor((int)pk[j][q], 16);
                }
            HU pb;
            #pragma unroll
            for (int q = 0; q < 4; ++q) {
                pb.u[q]     = hi ? xk[1][q] : pk[0][q];
                pb.u[4 + q] = hi ? pk[1][q] : xk[0][q];
            }

            // ---- O^T += V^T_tile @ P^T ----
            #pragma unroll
            for (int t = 0; t < 4; ++t) {
                v16h va;
                #pragma unroll
                for (int e = 0; e < 8; ++e) {
                    va[e]     = sVT[buf][16 * t + l16][16 * hi + e];
                    va[8 + e] = sVT[buf][16 * t + l16][16 * hi + 8 + e];
                }
                o[t] = __builtin_amdgcn_wmma_f32_16x16x32_f16(false, va, false, pb.h,
                                                              (short)0, o[t], false, false);
            }
        }
    }

    const float inv = 1.0f / lsum;
    float* orow = Ob + (size_t)myq * rowStride;
    #pragma unroll
    for (int t = 0; t < 4; ++t)
        #pragma unroll
        for (int r = 0; r < 8; ++r)
            orow[16 * t + 8 * hi + r] = o[t][r] * inv;
}

// =====================================================================
// Fallback fused kernel (round-2, f32 inputs, no workspace) — used only
// if d_ws is too small for the f16 staging buffers.
// =====================================================================
__global__ __launch_bounds__(NTHREADS)
void fa_causal_wmma_fused(const float* __restrict__ Qg,
                          const float* __restrict__ Kg,
                          const float* __restrict__ Vg,
                          float* __restrict__ Og)
{
    __shared__ __align__(16) _Float16 sK[BC][Ee + KPAD];
    __shared__ __align__(16) _Float16 sVT[Ee][BC + VPAD];

    const int b  = blockIdx.z;
    const int h  = blockIdx.y;
    const int q0 = blockIdx.x * BR;

    const int tid  = threadIdx.x;
    const int wave = tid >> 5;
    const int lane = tid & 31;
    const int l16  = lane & 15;
    const int hi   = lane >> 4;

    const size_t rowStride = (size_t)Hh * Ee;
    const size_t headOff   = ((size_t)b * Ll) * rowStride + (size_t)h * Ee;
    const float* Qb = Qg + headOff;
    const float* Kb = Kg + headOff;
    const float* Vb = Vg + headOff;
    float*       Ob = Og + headOff;

    const int qrow_base = q0 + wave * 16;
    const int myq       = qrow_base + l16;

    v16h qb[2];
    {
        const float* qrow = Qb + (size_t)myq * rowStride;
        #pragma unroll
        for (int c = 0; c < 2; ++c)
            #pragma unroll
            for (int e = 0; e < 16; ++e)
                qb[c][e] = (_Float16)qrow[32 * c + 8 * hi + (e & 7) + 16 * (e >> 3)];
    }

    v8f o[4] = {};
    float m = -INFINITY, lsum = 0.0f;

    const int jmax = (q0 + BR + BC - 1) / BC;

    for (int jt = 0; jt < jmax; ++jt) {
        const int k0 = jt * BC;
        __syncthreads();
        {
            const int r  = tid >> 3;
            const int cb = (tid & 7) * 8;
            const float* krow = Kb + (size_t)(k0 + r) * rowStride + cb;
            const float* vrow = Vb + (size_t)(k0 + r) * rowStride + cb;
            #pragma unroll
            for (int e = 0; e < 8; ++e) {
                sK[r][cb + e]  = (_Float16)krow[e];
                sVT[cb + e][r] = (_Float16)vrow[e];
            }
            if (jt + 1 < jmax) {
                __builtin_prefetch(krow + BC * rowStride, 0, 0);
                __builtin_prefetch(vrow + BC * rowStride, 0, 0);
            }
        }
        __syncthreads();

        if (__builtin_amdgcn_readfirstlane((int)(k0 <= qrow_base + 15))) {
            v8f s[2];
            #pragma unroll
            for (int j = 0; j < 2; ++j) {
                v16h ka0, ka1;
                #pragma unroll
                for (int e = 0; e < 8; ++e) {
                    ka0[e]     = sK[16 * j + l16][8 * hi + e];
                    ka0[8 + e] = sK[16 * j + l16][16 + 8 * hi + e];
                    ka1[e]     = sK[16 * j + l16][32 + 8 * hi + e];
                    ka1[8 + e] = sK[16 * j + l16][48 + 8 * hi + e];
                }
                v8f acc = {};
                acc = __builtin_amdgcn_wmma_f32_16x16x32_f16(false, ka0, false, qb[0],
                                                             (short)0, acc, false, false);
                acc = __builtin_amdgcn_wmma_f32_16x16x32_f16(false, ka1, false, qb[1],
                                                             (short)0, acc, false, false);
                s[j] = acc;
            }
            if (k0 + BC - 1 > qrow_base) {
                #pragma unroll
                for (int j = 0; j < 2; ++j)
                    #pragma unroll
                    for (int r = 0; r < 8; ++r)
                        if (k0 + 16 * j + 8 * hi + r > myq) s[j][r] = -INFINITY;
            }
            float rmax = fmaxf(s[0][0], s[1][0]);
            #pragma unroll
            for (int r = 1; r < 8; ++r)
                rmax = fmaxf(rmax, fmaxf(s[0][r], s[1][r]));
            rmax = fmaxf(rmax, __shfl_xor(rmax, 16));
            const float mnew = fmaxf(m, rmax);
            const float corr = __expf(m - mnew);
            m = mnew;
            float rs = 0.0f;
            #pragma unroll
            for (int j = 0; j < 2; ++j)
                #pragma unroll
                for (int r = 0; r < 8; ++r) {
                    s[j][r] = __expf(s[j][r] - mnew);
                    rs += s[j][r];
                }
            rs += __shfl_xor(rs, 16);
            lsum = lsum * corr + rs;
            #pragma unroll
            for (int t = 0; t < 4; ++t)
                #pragma unroll
                for (int r = 0; r < 8; ++r)
                    o[t][r] *= corr;

            unsigned pk[2][4], xk[2][4];
            #pragma unroll
            for (int j = 0; j < 2; ++j)
                #pragma unroll
                for (int q = 0; q < 4; ++q) {
                    v2h t;
                    t[0] = (_Float16)s[j][2 * q];
                    t[1] = (_Float16)s[j][2 * q + 1];
                    pk[j][q] = __builtin_bit_cast(unsigned, t);
                    xk[j][q] = (unsigned)__shfl_xor((int)pk[j][q], 16);
                }
            HU pb;
            #pragma unroll
            for (int q = 0; q < 4; ++q) {
                pb.u[q]     = hi ? xk[1][q] : pk[0][q];
                pb.u[4 + q] = hi ? pk[1][q] : xk[0][q];
            }
            #pragma unroll
            for (int t = 0; t < 4; ++t) {
                v16h va;
                #pragma unroll
                for (int e = 0; e < 8; ++e) {
                    va[e]     = sVT[16 * t + l16][16 * hi + e];
                    va[8 + e] = sVT[16 * t + l16][16 * hi + 8 + e];
                }
                o[t] = __builtin_amdgcn_wmma_f32_16x16x32_f16(false, va, false, pb.h,
                                                              (short)0, o[t], false, false);
            }
        }
    }

    const float inv = 1.0f / lsum;
    float* orow = Ob + (size_t)myq * rowStride;
    #pragma unroll
    for (int t = 0; t < 4; ++t)
        #pragma unroll
        for (int r = 0; r < 8; ++r)
            orow[16 * t + 8 * hi + r] = o[t][r] * inv;
}

extern "C" void kernel_launch(void* const* d_in, const int* in_sizes, int n_in,
                              void* d_out, int out_size, void* d_ws, size_t ws_size,
                              hipStream_t stream) {
    (void)in_sizes; (void)n_in; (void)out_size;
    const float* Q = (const float*)d_in[0];
    const float* K = (const float*)d_in[1];
    const float* V = (const float*)d_in[2];
    float*       O = (float*)d_out;

    const size_t elems  = (size_t)Bb * Hh * Ll * Ee;
    const size_t needWs = elems * sizeof(_Float16) * 2;   // Kh + VTh = 16 MB

    if (d_ws != nullptr && ws_size >= needWs) {
        _Float16* Kh  = (_Float16*)d_ws;
        _Float16* VTh = Kh + elems;
        cvt_kv_kernel<<<dim3(Ll / 64, Hh, Bb), 256, 0, stream>>>(K, V, Kh, VTh);
        fa_causal_wmma_split<<<dim3(Ll / BR, Hh, Bb), NTHREADS, 0, stream>>>(Q, Kh, VTh, O);
    } else {
        fa_causal_wmma_fused<<<dim3(Ll / BR, Hh, Bb), NTHREADS, 0, stream>>>(Q, K, V, O);
    }
}
